// NerfactoField_8263517078169
// MI455X (gfx1250) — matrix operand order repaired and checked by reading, not compiled, not run on hardware
//
#include <hip/hip_runtime.h>
#include <hip/hip_bf16.h>

// CDNA5 / gfx1250 fused NeRF-W field forward.
//  - prep_weights: f32 -> f16 transposed+padded weight pack into d_ws
//  - nerf_fused:   hash-encode + SH + 3 MLPs via v_wmma_f32_16x16x32_f16
// Wave32; each wave owns a 16-sample M-tile. Activations staged in LDS in the
// exact ISA A-matrix layout; weights read as B-fragments of two 16B chunks per lane.

typedef _Float16 h16;
typedef __attribute__((ext_vector_type(16))) _Float16 v16h;
typedef __attribute__((ext_vector_type(8)))  _Float16 v8h;
typedef __attribute__((ext_vector_type(8)))  float    v8f;

#define NTOT   262144
#define TSZ    (1u << 19)

// Workspace offsets (in halfs). Layout per weight: transposed [Npad][Kpad] f16.
#define OFF_B0 0        // w_base0: K=32  N=64   -> 32x64   = 2048
#define OFF_B1 2048     // w_base1: K=64  N=16   -> 64x16   = 1024
#define OFF_H0 3072     // w_head0: K=63->64 N=64-> 64x64   = 4096
#define OFF_H1 7168     // w_head1: K=64  N=64   -> 64x64   = 4096
#define OFF_H2 11264    // w_head2: K=64  N=3->16-> 64x16   = 1024
#define OFF_T0 12288    // w_tr0:   K=31->32 N=128->32x128  = 4096
#define OFF_T1 16384    // w_tr1:   K=128 N=128  -> 128x128 = 16384
#define OFF_T2 32768    // w_tr2:   K=128 N=5->16-> 128x16  = 2048
#define WS_HALFS 34816

// ---------- WMMA fragment helpers (ISA 7.12.2 layouts, wave32) ----------

// A 16x32 f16: lane l -> m = l&15, hi = l>>4; element e -> k = (e<8?0:16)+hi*8+(e&7)
__device__ inline v16h load_A(const h16* base, int Kpad, int kt, int m, int hi) {
  const h16* p = base + m * Kpad + kt * 32;
  v8h lo = *(const v8h*)(p + hi * 8);         // k = kt*32 + hi*8 .. +7
  v8h hh = *(const v8h*)(p + 16 + hi * 8);    // k = kt*32 + 16 + hi*8 .. +7
  return __builtin_shufflevector(lo, hh, 0,1,2,3,4,5,6,7,8,9,10,11,12,13,14,15);
}

// B 32x16 f16: lane l -> n = l&15, hi = l>>4; element e -> k = (e<8?0:16)+hi*8+(e&7)
// Weights stored transposed [n][k] so a fragment is two 8-half (16B) chunks.
__device__ inline v16h load_B(const h16* __restrict__ wT, int Kpad, int nt, int kt,
                              int n, int hi) {
  const h16* p = wT + (size_t)(nt * 16 + n) * Kpad + kt * 32;
  v8h lo = *(const v8h*)(p + hi * 8);
  v8h hh = *(const v8h*)(p + 16 + hi * 8);
  return __builtin_shufflevector(lo, hh, 0,1,2,3,4,5,6,7,8,9,10,11,12,13,14,15);
}

__device__ inline v8f wmma_f16(v16h a, v16h b, v8f c) {
  return __builtin_amdgcn_wmma_f32_16x16x32_f16(false, a, false, b, (short)0, c,
                                                false, false);
}

// C/D 16x16 f32: element v -> m = hi*8 + v, n = lane&15. Fused ReLU + f16 store.
__device__ inline void store_relu(h16* dst, int Kn, int nt, v8f c, int n, int hi) {
#pragma unroll
  for (int v = 0; v < 8; ++v) {
    float x = c[v];
    x = x > 0.f ? x : 0.f;
    dst[(hi * 8 + v) * Kn + nt * 16 + n] = (h16)x;
  }
}

// Wave-local LDS ordering: CDNA5 split counter wait (LDS ops are in-order).
__device__ inline void lds_fence() { asm volatile("s_wait_dscnt 0x0" ::: "memory"); }

// ---------- weight pack kernel ----------

__device__ inline void conv_wT(const float* __restrict__ src, h16* __restrict__ dst,
                               int K, int N, int Kpad, int Npad) {
  int total = Npad * Kpad;
  int stride = (int)(gridDim.x * blockDim.x);
  for (int i = (int)(blockIdx.x * blockDim.x + threadIdx.x); i < total; i += stride) {
    int n = i / Kpad, k = i - n * Kpad;
    float v = (k < K && n < N) ? src[(size_t)k * N + n] : 0.f;
    dst[i] = (h16)v;
  }
}

__global__ void prep_weights(const float* wb0, const float* wb1,
                             const float* wh0, const float* wh1, const float* wh2,
                             const float* wt0, const float* wt1, const float* wt2,
                             h16* ws) {
  conv_wT(wb0, ws + OFF_B0, 32, 64, 32, 64);
  conv_wT(wb1, ws + OFF_B1, 64, 16, 64, 16);
  conv_wT(wh0, ws + OFF_H0, 63, 64, 64, 64);
  conv_wT(wh1, ws + OFF_H1, 64, 64, 64, 64);
  conv_wT(wh2, ws + OFF_H2, 64, 3, 64, 16);
  conv_wT(wt0, ws + OFF_T0, 31, 128, 32, 128);
  conv_wT(wt1, ws + OFF_T1, 128, 128, 128, 128);
  conv_wT(wt2, ws + OFF_T2, 128, 5, 128, 16);
}

// ---------- fused field kernel ----------

__launch_bounds__(128)
__global__ void nerf_fused(const float* __restrict__ pos,
                           const float* __restrict__ dirs,
                           const float* __restrict__ app,
                           const float* __restrict__ trs,
                           const float* __restrict__ tables,
                           const h16* __restrict__ wsW,
                           float* __restrict__ out) {
  constexpr int WAVES = 4;
  __shared__ __align__(32) h16 sBufA[WAVES][16 * 128];
  __shared__ __align__(32) h16 sBufB[WAVES][16 * 128];
  __shared__ __align__(32) h16 sTrIn[WAVES][16 * 32];
  __shared__ float sSel[WAVES][16];

  const int w    = (int)threadIdx.x >> 5;
  const int lane = (int)threadIdx.x & 31;
  const int m    = lane & 15;   // sample row within tile; also n for B/C frags
  const int hi   = lane >> 4;   // half-wave
  const int tile = (int)blockIdx.x * WAVES + w;
  const int sBase = tile * 16;
  const int s = sBase + m;

  h16* bufA   = sBufA[w];             // [16][64] or [16][128] staging
  h16* bufB   = sBufB[w];
  h16* encA   = bufB;                 // [16][32] hash encoding (alias, dead early)
  h16* headIn = bufB + 16 * 32;       // [16][64]: SH(16)|geo(15)|app(32)|pad
  h16* trIn   = sTrIn[w];             // [16][32]: geo(15)|trans(16)|pad
  float* sel  = sSel[w];

  // Warm weight lines in L0/L2 (global_prefetch_b8); ~544B per lane covers 68KB.
  __builtin_prefetch(wsW + (size_t)threadIdx.x * 272, 0, 0);

  // ---- hash-grid encoding: lane handles sample m, levels [hi*8, hi*8+8) ----
  float px = pos[(size_t)s * 3 + 0], py = pos[(size_t)s * 3 + 1],
        pz = pos[(size_t)s * 3 + 2];
  px = (px + 2.f) * 0.25f; py = (py + 2.f) * 0.25f; pz = (pz + 2.f) * 0.25f;
  bool inb = (px >= 0.f && px <= 1.f && py >= 0.f && py <= 1.f &&
              pz >= 0.f && pz <= 1.f);
  float se = inb ? 1.f : 0.f;
  px *= se; py *= se; pz *= se;
  if (hi == 0) sel[m] = se;

  const float2* tab = (const float2*)tables;
  for (int j = 0; j < 8; ++j) {
    int lv = hi * 8 + j;
    // SCALES[lv] = 16 * (2048/16)^(lv/15) = 2^(4 + 7*lv/15)
    float scale = exp2f(4.f + (7.f / 15.f) * (float)lv);
    float xs = px * scale, ys = py * scale, zs = pz * scale;
    float fx = floorf(xs), fy = floorf(ys), fz = floorf(zs);
    float wx = xs - fx, wy = ys - fy, wz = zs - fz;
    int ix = (int)fx, iy = (int)fy, iz = (int)fz;
    float a0 = 0.f, a1 = 0.f;
#pragma unroll
    for (int c = 0; c < 8; ++c) {
      int ox = (c >> 2) & 1, oy = (c >> 1) & 1, oz = c & 1;
      unsigned hsh = (unsigned)(ix + ox)
                   ^ ((unsigned)(iy + oy) * 2654435761u)
                   ^ ((unsigned)(iz + oz) * 805459861u);
      hsh &= (TSZ - 1u);
      float2 f = tab[(size_t)lv * TSZ + hsh];
      float wgt = (ox ? wx : 1.f - wx) * (oy ? wy : 1.f - wy) * (oz ? wz : 1.f - wz);
      a0 += wgt * f.x; a1 += wgt * f.y;
    }
    encA[m * 32 + 2 * lv]     = (h16)a0;
    encA[m * 32 + 2 * lv + 1] = (h16)a1;
  }

  // ---- SH deg-4 (lanes of half 0) ----
  if (hi == 0) {
    float x = dirs[(size_t)s * 3 + 0], y = dirs[(size_t)s * 3 + 1],
          z = dirs[(size_t)s * 3 + 2];
    float x2 = x * x, y2 = y * y, z2 = z * z, xy = x * y, yz = y * z, xz = x * z;
    h16* d = headIn + m * 64;
    d[0]  = (h16)0.28209479177387814f;
    d[1]  = (h16)(-0.48860251190291987f * y);
    d[2]  = (h16)(0.48860251190291987f * z);
    d[3]  = (h16)(-0.48860251190291987f * x);
    d[4]  = (h16)(1.0925484305920792f * xy);
    d[5]  = (h16)(-1.0925484305920792f * yz);
    d[6]  = (h16)(0.94617469575756f * z2 - 0.31539156525252f);
    d[7]  = (h16)(-1.0925484305920792f * xz);
    d[8]  = (h16)(0.5462742152960396f * (x2 - y2));
    d[9]  = (h16)(0.5900435899266435f * y * (-3.f * x2 + y2));
    d[10] = (h16)(2.890611442640554f * xy * z);
    d[11] = (h16)(0.4570457994644657f * y * (1.f - 5.f * z2));
    d[12] = (h16)(0.3731763325901154f * z * (5.f * z2 - 3.f));
    d[13] = (h16)(0.4570457994644657f * x * (1.f - 5.f * z2));
    d[14] = (h16)(1.445305721320277f * z * (x2 - y2));
    d[15] = (h16)(0.5900435899266435f * x * (-x2 + 3.f * y2));
    d[63] = (h16)0.f;
    trIn[m * 32 + 31] = (h16)0.f;
  }
  // appearance embedding -> headIn cols 31..62 (split by half-wave)
#pragma unroll
  for (int j = 0; j < 16; ++j)
    headIn[m * 64 + 31 + hi * 16 + j] = (h16)app[(size_t)s * 32 + hi * 16 + j];
  // transient embedding -> trIn cols 15..30
#pragma unroll
  for (int j = 0; j < 8; ++j)
    trIn[m * 32 + 15 + hi * 8 + j] = (h16)trs[(size_t)s * 16 + hi * 8 + j];

  lds_fence();

  // ---- base MLP: enc[16x32] @ w_base0[32x64] -> relu -> @ w_base1[64x16] ----
  {
    v16h a0 = load_A(encA, 32, 0, m, hi);
#pragma unroll
    for (int nt = 0; nt < 4; ++nt) {
      v8f c = {};
      c = wmma_f16(a0, load_B(wsW + OFF_B0, 32, nt, 0, m, hi), c);
      store_relu(bufA, 64, nt, c, m, hi);
    }
  }
  lds_fence();
  {
    v8f c = {};
#pragma unroll
    for (int kt = 0; kt < 2; ++kt) {
      v16h a = load_A(bufA, 64, kt, m, hi);
      c = wmma_f16(a, load_B(wsW + OFF_B1, 64, 0, kt, m, hi), c);
    }
    // col 0: density = trunc_exp(raw)*sel; cols 1..15: geo -> head/transient in
#pragma unroll
    for (int v = 0; v < 8; ++v) {
      int mm = hi * 8 + v;
      float val = c[v];
      if (m == 0) {
        out[3 * NTOT + sBase + mm] = expf(val) * sel[mm];
      } else {
        h16 g = (h16)val;
        headIn[mm * 64 + 16 + (m - 1)] = g;
        trIn[mm * 32 + (m - 1)] = g;
      }
    }
  }
  lds_fence();

  // ---- head MLP: [16x64] -> 64 -> 64 -> 3 (sigmoid) ----
  {
    v16h a0 = load_A(headIn, 64, 0, m, hi);
    v16h a1 = load_A(headIn, 64, 1, m, hi);
#pragma unroll
    for (int nt = 0; nt < 4; ++nt) {
      v8f c = {};
      c = wmma_f16(a0, load_B(wsW + OFF_H0, 64, nt, 0, m, hi), c);
      c = wmma_f16(a1, load_B(wsW + OFF_H0, 64, nt, 1, m, hi), c);
      store_relu(bufA, 64, nt, c, m, hi);
    }
  }
  lds_fence();
  {
    v16h a0 = load_A(bufA, 64, 0, m, hi);
    v16h a1 = load_A(bufA, 64, 1, m, hi);
#pragma unroll
    for (int nt = 0; nt < 4; ++nt) {
      v8f c = {};
      c = wmma_f16(a0, load_B(wsW + OFF_H1, 64, nt, 0, m, hi), c);
      c = wmma_f16(a1, load_B(wsW + OFF_H1, 64, nt, 1, m, hi), c);
      store_relu(bufB, 64, nt, c, m, hi);
    }
  }
  lds_fence();
  {
    v16h a0 = load_A(bufB, 64, 0, m, hi);
    v16h a1 = load_A(bufB, 64, 1, m, hi);
    v8f c = {};
    c = wmma_f16(a0, load_B(wsW + OFF_H2, 64, 0, 0, m, hi), c);
    c = wmma_f16(a1, load_B(wsW + OFF_H2, 64, 0, 1, m, hi), c);
    if (m < 3) {
#pragma unroll
      for (int v = 0; v < 8; ++v) {
        int mm = hi * 8 + v;
        out[(size_t)(sBase + mm) * 3 + m] = 1.f / (1.f + expf(-c[v]));
      }
    }
  }

  // ---- transient MLP: [16x32] -> 128 -> 128 -> 5 ----
  {
    v16h a0 = load_A(trIn, 32, 0, m, hi);
#pragma unroll
    for (int nt = 0; nt < 8; ++nt) {
      v8f c = {};
      c = wmma_f16(a0, load_B(wsW + OFF_T0, 32, nt, 0, m, hi), c);
      store_relu(bufA, 128, nt, c, m, hi);
    }
  }
  lds_fence();
  {
    v16h a0 = load_A(bufA, 128, 0, m, hi);
    v16h a1 = load_A(bufA, 128, 1, m, hi);
    v16h a2 = load_A(bufA, 128, 2, m, hi);
    v16h a3 = load_A(bufA, 128, 3, m, hi);
#pragma unroll
    for (int nt = 0; nt < 8; ++nt) {
      v8f c = {};
      c = wmma_f16(a0, load_B(wsW + OFF_T1, 128, nt, 0, m, hi), c);
      c = wmma_f16(a1, load_B(wsW + OFF_T1, 128, nt, 1, m, hi), c);
      c = wmma_f16(a2, load_B(wsW + OFF_T1, 128, nt, 2, m, hi), c);
      c = wmma_f16(a3, load_B(wsW + OFF_T1, 128, nt, 3, m, hi), c);
      store_relu(bufB, 128, nt, c, m, hi);
    }
  }
  lds_fence();
  {
    v16h a0 = load_A(bufB, 128, 0, m, hi);
    v16h a1 = load_A(bufB, 128, 1, m, hi);
    v16h a2 = load_A(bufB, 128, 2, m, hi);
    v16h a3 = load_A(bufB, 128, 3, m, hi);
    v8f c = {};
    c = wmma_f16(a0, load_B(wsW + OFF_T2, 128, 0, 0, m, hi), c);
    c = wmma_f16(a1, load_B(wsW + OFF_T2, 128, 0, 1, m, hi), c);
    c = wmma_f16(a2, load_B(wsW + OFF_T2, 128, 0, 2, m, hi), c);
    c = wmma_f16(a3, load_B(wsW + OFF_T2, 128, 0, 3, m, hi), c);
#pragma unroll
    for (int v = 0; v < 8; ++v) {
      int mm = hi * 8 + v;
      int sIdx = sBase + mm;
      float val = c[v];
      if (m == 0) {
        out[4 * NTOT + sIdx] = expf(val) * sel[mm];            // density_transient
      } else if (m < 4) {
        out[5 * NTOT + (size_t)sIdx * 3 + (m - 1)] =
            1.f / (1.f + expf(-val));                          // rgb_transient
      } else if (m == 4) {
        out[8 * NTOT + sIdx] =
            (val > 20.f) ? val : log1pf(expf(val));            // uncertainty
      }
    }
  }
}

extern "C" void kernel_launch(void* const* d_in, const int* in_sizes, int n_in,
                              void* d_out, int out_size, void* d_ws, size_t ws_size,
                              hipStream_t stream) {
  (void)in_sizes; (void)n_in; (void)out_size; (void)ws_size;
  const float* positions = (const float*)d_in[0];
  const float* viewdirs  = (const float*)d_in[1];
  const float* app       = (const float*)d_in[2];
  const float* trs       = (const float*)d_in[3];
  const float* tables    = (const float*)d_in[4];
  const float* wb0 = (const float*)d_in[5];
  const float* wb1 = (const float*)d_in[6];
  const float* wh0 = (const float*)d_in[7];
  const float* wh1 = (const float*)d_in[8];
  const float* wh2 = (const float*)d_in[9];
  const float* wt0 = (const float*)d_in[10];
  const float* wt1 = (const float*)d_in[11];
  const float* wt2 = (const float*)d_in[12];
  h16* wsW = (h16*)d_ws;
  float* out = (float*)d_out;

  prep_weights<<<64, 256, 0, stream>>>(wb0, wb1, wh0, wh1, wh2, wt0, wt1, wt2, wsW);

  constexpr int WAVES = 4;
  int blocks = NTOT / (WAVES * 16);  // 4096
  nerf_fused<<<blocks, WAVES * 32, 0, stream>>>(positions, viewdirs, app, trs,
                                                tables, wsW, out);
}
